// AttentionLayer_81990925681016
// MI455X (gfx1250) — matrix operand order repaired and checked by reading, not compiled, not run on hardware
//
#include <hip/hip_runtime.h>

typedef __attribute__((ext_vector_type(16))) __bf16 v16bf;
typedef __attribute__((ext_vector_type(8)))  __bf16 v8bf;
typedef __attribute__((ext_vector_type(8)))  float  v8f;

#define B_    8
#define NF    512
#define LEN   2500
#define AD    256
#define CDIM  8921
#define LXP   2528            // padded L stride for x bf16 (79 * 32)
#define LPAD  2560            // padded L rows for xT / ZT (40 * 64)
#define LP    2528            // padded L stride in LDS
#define NT2   157             // ceil(2500/16) l-tiles in GEMM2
#define KT3   79              // 2528/32 k-steps in GEMM3
#define CT    558             // ceil(8921/16) c-tiles
#define CPAD  8928            // CT*16 padded u rows

#if __has_builtin(__builtin_amdgcn_sched_barrier)
#define SCHED_FENCE() __builtin_amdgcn_sched_barrier(0)
#else
#define SCHED_FENCE()
#endif

union ABf { v16bf v; v8bf h[2]; __bf16 e[16]; };

// ---------------------------------------------------------------- converts
__global__ void conv_x(const float* __restrict__ x,
                       __bf16* __restrict__ xbf,   // [B][NF][LXP]
                       __bf16* __restrict__ xT)    // [B][LPAD][NF]
{
    const int row = blockIdx.x;             // b*NF + f
    const int b = row / NF, f = row % NF;
    for (int l = threadIdx.x; l < LXP; l += blockDim.x) {
        if (l < LEN) {
            float v = x[(size_t)row * LEN + l];
            xbf[(size_t)row * LXP + l] = (__bf16)v;
            xT[((size_t)b * LPAD + l) * NF + f] = (__bf16)v;
        } else {
            xbf[(size_t)row * LXP + l] = (__bf16)0.0f;   // zero pad
        }
    }
}

__global__ void conv_flat(const float* __restrict__ src, __bf16* __restrict__ dst, int n)
{
    int i = blockIdx.x * blockDim.x + threadIdx.x;
    int stride = gridDim.x * blockDim.x;
    for (; i < n; i += stride) dst[i] = (__bf16)src[i];
}

// zero the pad rows of xT: per batch, rows LEN..LPAD-1  (60*512 elems each)
__global__ void pad_xT(__bf16* __restrict__ xT)
{
    const int chunk = (LPAD - LEN) * NF;                 // 30720
    int i = blockIdx.x * blockDim.x + threadIdx.x;
    int stride = gridDim.x * blockDim.x;
    for (; i < B_ * chunk; i += stride) {
        int b = i / chunk, r = i % chunk;
        xT[((size_t)b * LPAD + LEN) * NF + r] = (__bf16)0.0f;
    }
}

// zero pad rows of u: rows CDIM..CPAD-1
__global__ void pad_u(__bf16* __restrict__ ubf)
{
    int i = blockIdx.x * blockDim.x + threadIdx.x;
    int n = (CPAD - CDIM) * AD;                          // 1792
    if (i < n) ubf[(size_t)CDIM * AD + i] = (__bf16)0.0f;
}

// -------------------------------------------- GEMM1: Z^T = tanh(W * X)^T
// One wave computes a 16(d) x 64(l) strip. A = W tile, B = X (via xT).
// Unguarded: xT pad rows are zero, so ZT pad rows get tanh(0)=0.
__global__ void __launch_bounds__(256) gemm1_zt(const __bf16* __restrict__ wbf,  // [AD][NF]
                                                const __bf16* __restrict__ xT,   // [B][LPAD][NF]
                                                __bf16* __restrict__ ZT)         // [B][LPAD][AD]
{
    const int lane = threadIdx.x & 31;
    const int wid  = threadIdx.x >> 5;
    const int NLS  = 40;                          // l-strips of 64
    int task = blockIdx.x * 8 + wid;              // 0 .. 5119
    if (task >= B_ * 16 * NLS) return;
    const int b   = task / (16 * NLS);
    const int rr  = task % (16 * NLS);
    const int d0  = (rr / NLS) * 16;
    const int l0b = (rr % NLS) * 64;
    const int hl  = lane >> 4;
    const int mn  = lane & 15;

    v8f zero8 = {0.f,0.f,0.f,0.f,0.f,0.f,0.f,0.f};
    v8f acc[4] = {zero8, zero8, zero8, zero8};

    const __bf16* wrow  = wbf + (size_t)(d0 + mn) * NF;
    const __bf16* xrow0 = xT + ((size_t)b * LPAD + l0b + mn) * NF;

#define G1_LOAD(Af, Bf, K0)                                                     \
    Af.h[0] = *(const v8bf*)(wrow + (K0) + hl * 16);                            \
    Af.h[1] = *(const v8bf*)(wrow + (K0) + hl * 16 + 8);                        \
    _Pragma("unroll")                                                           \
    for (int t = 0; t < 4; ++t)                                                 \
        Bf[t].v = *(const v16bf*)(xrow0 + (size_t)t * 16 * NF + (K0) + hl * 16);

#define G1_WMMA(Af, Bf)                                                         \
    _Pragma("unroll")                                                           \
    for (int t = 0; t < 4; ++t)                                                 \
        acc[t] = __builtin_amdgcn_wmma_f32_16x16x32_bf16(                       \
            false, Af.v, false, Bf[t].v, (short)0, acc[t], false, false);

    ABf a0, a1, b0[4], b1[4];
    G1_LOAD(a0, b0, 0)
    for (int kk = 0; kk + 2 < 16; kk += 2) {      // steps 0..13, preloads up to 14
        G1_LOAD(a1, b1, (kk + 1) * 32)
        SCHED_FENCE();
        G1_WMMA(a0, b0)
        SCHED_FENCE();
        G1_LOAD(a0, b0, (kk + 2) * 32)
        SCHED_FENCE();
        G1_WMMA(a1, b1)
        SCHED_FENCE();
    }
    G1_LOAD(a1, b1, 15 * 32)                      // step 15
    SCHED_FENCE();
    G1_WMMA(a0, b0)                               // step 14
    G1_WMMA(a1, b1)                               // step 15
#undef G1_LOAD
#undef G1_WMMA

#pragma unroll
    for (int t = 0; t < 4; ++t) {
        const int l = l0b + t * 16 + mn;
        union { v8bf v; __bf16 e[8]; } o;
#pragma unroll
        for (int i = 0; i < 8; ++i) o.e[i] = (__bf16)tanhf(acc[t][i]);
        *(v8bf*)(ZT + ((size_t)b * LPAD + l) * AD + d0 + hl * 8) = o.v;
    }
}

// ---------------- fused: S = U*Z  ->  softmax_l  ->  V = E * X^T  (per 16-c tile)
__global__ void __launch_bounds__(256) fused_attn(const __bf16* __restrict__ ubf, // [CPAD][AD]
                                                  const __bf16* __restrict__ ZT,  // [B][LPAD][AD]
                                                  const __bf16* __restrict__ xbf, // [B][NF][LXP]
                                                  float* __restrict__ V)          // [B][C][NF]
{
    extern __shared__ char smem[];
    float*  Ssm  = (float*)smem;                                  // 16*LP f32
    __bf16* Esm  = (__bf16*)(smem + (size_t)16 * LP * 4);         // 16*LP bf16
    float*  red  = (float*)(smem + (size_t)16 * LP * 4 + (size_t)16 * LP * 2); // 256
    float*  rmax = red + 256;                                     // 16
    float*  rsmi = rmax + 16;                                     // 16

    const int tid  = threadIdx.x;
    const int lane = tid & 31;
    const int wid  = tid >> 5;
    const int hl   = lane >> 4;
    const int mn   = lane & 15;

    const int b  = blockIdx.x / CT;
    const int c0 = (blockIdx.x % CT) * 16;

    v8f zero8 = {0.f,0.f,0.f,0.f,0.f,0.f,0.f,0.f};

    // hoisted U A-fragments for all 8 K-steps (K = 256); pad rows are zero
    ABf ua[8];
    {
        const __bf16* urow = ubf + (size_t)(c0 + mn) * AD;
#pragma unroll
        for (int kk = 0; kk < 8; ++kk) {
            ua[kk].h[0] = *(const v8bf*)(urow + kk * 32 + hl * 16);
            ua[kk].h[1] = *(const v8bf*)(urow + kk * 32 + hl * 16 + 8);
        }
    }

    // ---- Phase A: S tile rows into LDS (ZT pad rows are zero) ----
    for (int t = wid; t < NT2; t += 8) {
        const int l0 = t * 16;
        const __bf16* zrow = ZT + ((size_t)b * LPAD + l0 + mn) * AD;
        v8f acc = zero8;
        ABf pb0, pb1;
        pb0.v = *(const v16bf*)(zrow + hl * 16);
#pragma unroll
        for (int kk = 0; kk < 6; kk += 2) {
            pb1.v = *(const v16bf*)(zrow + (kk + 1) * 32 + hl * 16);
            SCHED_FENCE();
            acc = __builtin_amdgcn_wmma_f32_16x16x32_bf16(
                false, ua[kk].v, false, pb0.v, (short)0, acc, false, false);
            SCHED_FENCE();
            pb0.v = *(const v16bf*)(zrow + (kk + 2) * 32 + hl * 16);
            SCHED_FENCE();
            acc = __builtin_amdgcn_wmma_f32_16x16x32_bf16(
                false, ua[kk + 1].v, false, pb1.v, (short)0, acc, false, false);
            SCHED_FENCE();
        }
        pb1.v = *(const v16bf*)(zrow + 7 * 32 + hl * 16);
        SCHED_FENCE();
        acc = __builtin_amdgcn_wmma_f32_16x16x32_bf16(
            false, ua[6].v, false, pb0.v, (short)0, acc, false, false);
        acc = __builtin_amdgcn_wmma_f32_16x16x32_bf16(
            false, ua[7].v, false, pb1.v, (short)0, acc, false, false);
#pragma unroll
        for (int i = 0; i < 8; ++i)
            Ssm[(size_t)(i + hl * 8) * LP + l0 + mn] = acc[i];
    }
    __syncthreads();

    // ---- Phase B: row max, exp -> bf16 E, 1/sum ----
    {
        const int r = tid >> 4;       // 0..15 : c-row in tile
        const int j = tid & 15;
        float m = -3.4e38f;
        for (int l = j; l < LEN; l += 16) m = fmaxf(m, Ssm[(size_t)r * LP + l]);
        red[tid] = m;
        __syncthreads();
        if (j == 0) {
            float mm = red[r * 16];
            for (int q = 1; q < 16; ++q) mm = fmaxf(mm, red[r * 16 + q]);
            rmax[r] = mm;
        }
        __syncthreads();
        const float rm = rmax[r];
        float s = 0.f;
        for (int l = j; l < LEN; l += 16) {
            float e = __expf(Ssm[(size_t)r * LP + l] - rm);
            s += e;
            Esm[(size_t)r * LP + l] = (__bf16)e;
        }
        for (int l = LEN + j; l < LP; l += 16)
            Esm[(size_t)r * LP + l] = (__bf16)0.0f;      // zero K-pad
        red[tid] = s;
        __syncthreads();
        if (j == 0) {
            float ss = 0.f;
            for (int q = 0; q < 16; ++q) ss += red[r * 16 + q];
            rsmi[r] = 1.f / ss;
        }
        __syncthreads();
    }

    // ---- Phase C: V tile = E * X^T, each wave owns 64 f-columns ----
    const int f0b = wid * 64;
    v8f acc[4] = {zero8, zero8, zero8, zero8};
    const __bf16* xrow0 = xbf + ((size_t)b * NF + f0b + mn) * LXP;
    const __bf16* erow  = Esm + (size_t)mn * LP;

#define PC_LOAD(Af, Bf, K0)                                                       \
    Af.h[0] = *(const v8bf*)(erow + (K0) + hl * 16);                              \
    Af.h[1] = *(const v8bf*)(erow + (K0) + hl * 16 + 8);                          \
    _Pragma("unroll")                                                             \
    for (int t = 0; t < 4; ++t)                                                   \
        Bf[t].v = *(const v16bf*)(xrow0 + (size_t)t * 16 * LXP + (K0) + hl * 16);

#define PC_WMMA(Af, Bf)                                                           \
    _Pragma("unroll")                                                             \
    for (int t = 0; t < 4; ++t)                                                   \
        acc[t] = __builtin_amdgcn_wmma_f32_16x16x32_bf16(                         \
            false, Af.v, false, Bf[t].v, (short)0, acc[t], false, false);

    {
        ABf a0, a1, b0[4], b1[4];
        PC_LOAD(a0, b0, 0)
        for (int kk = 0; kk + 2 < KT3; kk += 2) {   // steps 0..77, preloads up to 78
            PC_LOAD(a1, b1, (kk + 1) * 32)
            SCHED_FENCE();
            PC_WMMA(a0, b0)
            SCHED_FENCE();
            PC_LOAD(a0, b0, (kk + 2) * 32)
            SCHED_FENCE();
            PC_WMMA(a1, b1)
            SCHED_FENCE();
        }
        PC_WMMA(a0, b0)                             // step 78 (KT3-1, odd count)
    }
#undef PC_LOAD
#undef PC_WMMA

#pragma unroll
    for (int t = 0; t < 4; ++t) {
        const int f = f0b + t * 16 + mn;
#pragma unroll
        for (int i = 0; i < 8; ++i) {
            const int cr = i + hl * 8;
            const int c  = c0 + cr;
            if (c < CDIM)
                V[((size_t)b * CDIM + c) * NF + f] = acc[t][i] * rsmi[cr];
        }
    }
}

// ---------------------------------------------------------------- launch
extern "C" void kernel_launch(void* const* d_in, const int* in_sizes, int n_in,
                              void* d_out, int out_size, void* d_ws, size_t ws_size,
                              hipStream_t stream)
{
    (void)in_sizes; (void)n_in; (void)out_size; (void)ws_size;
    const float* x = (const float*)d_in[0];
    const float* w = (const float*)d_in[1];
    const float* u = (const float*)d_in[2];
    float* V = (float*)d_out;

    char* ws = (char*)d_ws;
    size_t off = 0;
    __bf16* xbf = (__bf16*)(ws + off); off += (size_t)B_ * NF * LXP * 2;    // 20.7 MB
    __bf16* xT  = (__bf16*)(ws + off); off += (size_t)B_ * LPAD * NF * 2;   // 21.0 MB
    __bf16* wbf = (__bf16*)(ws + off); off += (size_t)AD * NF * 2;          // 256 KB
    __bf16* ubf = (__bf16*)(ws + off); off += (size_t)CPAD * AD * 2;        // 4.6 MB
    __bf16* ZT  = (__bf16*)(ws + off); off += (size_t)B_ * LPAD * AD * 2;   // 10.5 MB

    conv_x<<<B_ * NF, 256, 0, stream>>>(x, xbf, xT);
    pad_xT<<<240, 256, 0, stream>>>(xT);
    conv_flat<<<512, 256, 0, stream>>>(w, wbf, AD * NF);
    conv_flat<<<2048, 256, 0, stream>>>(u, ubf, CDIM * AD);
    pad_u<<<7, 256, 0, stream>>>(ubf);

    gemm1_zt<<<640, 256, 0, stream>>>(wbf, xT, ZT);

    const size_t lds = (size_t)16 * LP * 4 + (size_t)16 * LP * 2 + 256 * 4 + 32 * 4; // 243,840 B
    fused_attn<<<B_ * CT, 256, lds, stream>>>(ubf, ZT, xbf, V);
}
